// SimpleMambaClassifierLuenberger_14353780703722
// MI455X (gfx1250) — matrix operand order repaired and checked by reading, not compiled, not run on hardware
//
#include <hip/hip_runtime.h>
#include <hip/hip_bf16.h>

// ---------------------------------------------------------------------------
// SimpleMambaClassifierLuenberger for MI455X (gfx1250, wave32, WMMA)
// B=32, L=1024, DM=128, DI=256, DS=16, DC=4, DR=8, NC=10, NL=4, ALPHA=0.3
// ---------------------------------------------------------------------------

#define NLAY 4
#define DM   128
#define DI   256
#define DSZ  16
#define DCV  4
#define DRK  8
#define NCLS 10
#define BATCH 32
#define LSEQ 1024
#define BL   (BATCH * LSEQ)   // 32768 tokens

typedef __attribute__((ext_vector_type(16))) _Float16 v16h;
typedef __attribute__((ext_vector_type(8)))  _Float16 v8h;
typedef __attribute__((ext_vector_type(8)))  float    v8f;

// ---------------------------------------------------------------------------
// Pack weights W[layer][K][N] (row-major f32) into per-tile f16 B-matrix lane
// layout for v_wmma_f32_16x16x32_f16:
//   tile (nt,kt), lane l (0..31), elem i (0..15):
//     column n = nt*16 + (l&15)
//     k        = kt*32 + i + (l<16 ? 0 : 16)
//   packed[(((layer*NT + nt)*KT + kt)*32 + l)*16 + i]
// ---------------------------------------------------------------------------
__global__ void pack_w_kernel(const float* __restrict__ W, _Float16* __restrict__ P,
                              int K, int N, int total) {
  int idx = blockIdx.x * blockDim.x + threadIdx.x;
  if (idx >= total) return;
  int perLayer = K * N;
  int layer = idx / perLayer;
  int r = idx - layer * perLayer;
  int i = r & 15;
  int l = (r >> 4) & 31;
  int tile = r >> 9;                 // nt * (K/32) + kt
  int ktiles = K >> 5;
  int nt = tile / ktiles;
  int kt = tile - nt * ktiles;
  int n = nt * 16 + (l & 15);
  int k = kt * 32 + i + ((l < 16) ? 0 : 16);
  P[idx] = (_Float16)W[(size_t)layer * perLayer + (size_t)k * N + n];
}

// ---------------------------------------------------------------------------
// Embed: h[t, m] = sum_c x[b, c, l] * W_in[c, m] + b_in[m]   (t = b*L + l)
// ---------------------------------------------------------------------------
__global__ void embed_kernel(const float* __restrict__ x, const float* __restrict__ Win,
                             const float* __restrict__ bin, float* __restrict__ h) {
  int idx = blockIdx.x * blockDim.x + threadIdx.x;   // t*DM + m
  if (idx >= BL * DM) return;
  int m = idx & (DM - 1);
  int t = idx >> 7;
  int b = t >> 10;
  int l = t & (LSEQ - 1);
  const float* xb = x + (size_t)b * 3 * LSEQ;
  float acc = bin[m];
#pragma unroll
  for (int c = 0; c < 3; ++c) acc += xb[c * LSEQ + l] * Win[c * DM + m];
  h[idx] = acc;
}

// ---------------------------------------------------------------------------
// LayerNorm(128) per token -> f16.  One wave32 per token, 4 elems per lane.
// ---------------------------------------------------------------------------
__global__ void ln_to_f16_kernel(const float* __restrict__ h, const float* __restrict__ w,
                                 const float* __restrict__ bb, _Float16* __restrict__ z) {
  int gtid = blockIdx.x * blockDim.x + threadIdx.x;
  int t = gtid >> 5;
  int lane = threadIdx.x & 31;
  if (t >= BL) return;
  const float4 xv = reinterpret_cast<const float4*>(h + (size_t)t * DM)[lane];
  float s = xv.x + xv.y + xv.z + xv.w;
#pragma unroll
  for (int off = 16; off; off >>= 1) s += __shfl_xor(s, off, 32);
  float m = s * (1.0f / DM);
  float d0 = xv.x - m, d1 = xv.y - m, d2 = xv.z - m, d3 = xv.w - m;
  float v = d0 * d0 + d1 * d1 + d2 * d2 + d3 * d3;
#pragma unroll
  for (int off = 16; off; off >>= 1) v += __shfl_xor(v, off, 32);
  float rs = rsqrtf(v * (1.0f / DM) + 1e-5f);
  const float4 wv = reinterpret_cast<const float4*>(w)[lane];
  const float4 bv = reinterpret_cast<const float4*>(bb)[lane];
  _Float16* zo = z + (size_t)t * DM + lane * 4;
  zo[0] = (_Float16)(d0 * rs * wv.x + bv.x);
  zo[1] = (_Float16)(d1 * rs * wv.y + bv.y);
  zo[2] = (_Float16)(d2 * rs * wv.z + bv.z);
  zo[3] = (_Float16)(d3 * rs * wv.w + bv.w);
}

// ---------------------------------------------------------------------------
// WMMA GEMM: C[M x N] = A16[M x K] @ packedW.  One 16x64 tile per wave32
// (4 independent accumulator chains -> A fragment reused 4x, WMMA RAW hazard
// hidden by chain interleave).  N must be a multiple of 64.
// fuse==1: C = res + acc + bias[col]  (out_proj epilogue with residual)
// ---------------------------------------------------------------------------
__global__ void wmma_gemm_kernel(const _Float16* __restrict__ A, const _Float16* __restrict__ P,
                                 float* __restrict__ C, const float* __restrict__ res,
                                 const float* __restrict__ bias,
                                 int M, int N, int K, int fuse) {
  int wavesPerBlock = blockDim.x >> 5;
  int wave = threadIdx.x >> 5;
  int lane = threadIdx.x & 31;
  int tileId = blockIdx.x * wavesPerBlock + wave;       // wave-uniform
  int nBlks = N >> 6;                                    // 64-wide column blocks
  int mt = tileId / nBlks;
  int nb = tileId - mt * nBlks;
  if (mt >= (M >> 4)) return;                           // whole-wave uniform exit

  int row = mt * 16 + (lane & 15);
  int khalf = (lane < 16) ? 0 : 16;
  int ktiles = K >> 5;
  const _Float16* aRow = A + (size_t)row * K;
  // B tiles for nt = nb*4 + j, each tile = 512 halves
  const _Float16* bBase = P + (size_t)(nb * 4 * ktiles) * 512 + lane * 16;

  v8f acc0 = {}, acc1 = {}, acc2 = {}, acc3 = {};
  for (int kt = 0; kt < ktiles; ++kt) {
    // prefetch A stream ahead (global_prefetch_b8)
    __builtin_prefetch(aRow + kt * 32 + 128, 0, 0);
    v8h a0 = *reinterpret_cast<const v8h*>(aRow + kt * 32 + khalf);
    v8h a1 = *reinterpret_cast<const v8h*>(aRow + kt * 32 + 8 + khalf);
    v16h a;
#pragma unroll
    for (int i = 0; i < 8; ++i) { a[i] = a0[i]; a[i + 8] = a1[i]; }
    v16h b0 = *reinterpret_cast<const v16h*>(bBase + (size_t)(0 * ktiles + kt) * 512);
    v16h b1 = *reinterpret_cast<const v16h*>(bBase + (size_t)(1 * ktiles + kt) * 512);
    v16h b2 = *reinterpret_cast<const v16h*>(bBase + (size_t)(2 * ktiles + kt) * 512);
    v16h b3 = *reinterpret_cast<const v16h*>(bBase + (size_t)(3 * ktiles + kt) * 512);
    acc0 = __builtin_amdgcn_wmma_f32_16x16x32_f16(false, a, false, b0, (short)0, acc0, false, false);
    acc1 = __builtin_amdgcn_wmma_f32_16x16x32_f16(false, a, false, b1, (short)0, acc1, false, false);
    acc2 = __builtin_amdgcn_wmma_f32_16x16x32_f16(false, a, false, b2, (short)0, acc2, false, false);
    acc3 = __builtin_amdgcn_wmma_f32_16x16x32_f16(false, a, false, b3, (short)0, acc3, false, false);
  }

  int rbase = mt * 16 + ((lane < 16) ? 0 : 8);
  v8f accs[4] = {acc0, acc1, acc2, acc3};
#pragma unroll
  for (int j = 0; j < 4; ++j) {
    int col = (nb * 4 + j) * 16 + (lane & 15);
    float bj = fuse ? bias[col] : 0.f;
#pragma unroll
    for (int r = 0; r < 8; ++r) {
      size_t idx = (size_t)(rbase + r) * N + col;
      float vv = accs[j][r];
      if (fuse) vv += res[idx] + bj;
      C[idx] = vv;
    }
  }
}

// ---------------------------------------------------------------------------
// Causal depthwise conv (DC=4) + SiLU:  xc[t,d]
// ---------------------------------------------------------------------------
__global__ void conv_silu_kernel(const float* __restrict__ xz, const float* __restrict__ cw,
                                 const float* __restrict__ cb, float* __restrict__ xc) {
  int idx = blockIdx.x * blockDim.x + threadIdx.x;     // t*DI + d
  if (idx >= BL * DI) return;
  int d = idx & (DI - 1);
  int t = idx >> 8;
  int l = t & (LSEQ - 1);
  float acc = cb[d];
#pragma unroll
  for (int j = 0; j < DCV; ++j) {
    int ll = l - (DCV - 1) + j;
    if (ll >= 0) acc += xz[(size_t)(t - (DCV - 1) + j) * (2 * DI) + d] * cw[d * DCV + j];
  }
  xc[idx] = acc / (1.0f + __expf(-acc));
}

// ---------------------------------------------------------------------------
// x_proj: proj[t, o] = xc[t,:] @ x_proj_w[:, o],  o < 40  (K=256, tiny N)
// ---------------------------------------------------------------------------
__global__ void xproj_kernel(const float* __restrict__ xc, const float* __restrict__ W,
                             float* __restrict__ proj) {
  int idx = blockIdx.x * blockDim.x + threadIdx.x;     // t*40 + o
  if (idx >= BL * (DRK + 2 * DSZ)) return;
  const int NO = DRK + 2 * DSZ;                         // 40
  int o = idx % NO;
  int t = idx / NO;
  const float* x = xc + (size_t)t * DI;
  float acc = 0.f;
  for (int k = 0; k < DI; ++k) acc += x[k] * W[k * NO + o];
  proj[idx] = acc;
}

// ---------------------------------------------------------------------------
// delta[t,d] = softplus(dt[t,:] @ dt_proj_w[:,d] + dt_proj_b[d])   (K=8)
// ---------------------------------------------------------------------------
__global__ void delta_kernel(const float* __restrict__ proj, const float* __restrict__ Wdt,
                             const float* __restrict__ bdt, float* __restrict__ delta) {
  int idx = blockIdx.x * blockDim.x + threadIdx.x;     // t*DI + d
  if (idx >= BL * DI) return;
  int d = idx & (DI - 1);
  int t = idx >> 8;
  const float* dt = proj + (size_t)t * (DRK + 2 * DSZ);
  float acc = bdt[d];
#pragma unroll
  for (int r = 0; r < DRK; ++r) acc += dt[r] * Wdt[r * DI + d];
  delta[idx] = (acc > 20.f) ? acc : __logf(1.f + __expf(acc));
}

// ---------------------------------------------------------------------------
// Selective scan, barrier-free: each 16-lane group owns one (batch, channel);
// lane s holds state h[s].  y_t = sum_s h_s*C_s via 4 shfl_xor hops (stay
// inside the 16-lane group).  4096 independent wave32s (16x the parallelism
// of a block-per-batch design), no __syncthreads on the critical path.
// Fused: Dp skip, Luenberger blend (0.7/0.3), SiLU gate, f16 emit.
// ---------------------------------------------------------------------------
__global__ void ssm_kernel(const float* __restrict__ delta, const float* __restrict__ proj,
                           const float* __restrict__ xc, const float* __restrict__ xz,
                           const float* __restrict__ A_log, const float* __restrict__ Dp,
                           _Float16* __restrict__ y16) {
  const int NO = DRK + 2 * DSZ;                        // 40
  int wid = blockIdx.x * (blockDim.x >> 5) + (threadIdx.x >> 5);  // 0..4095
  int lane = threadIdx.x & 31;
  int s = lane & 15;
  int b = wid >> 7;                                    // 128 waves per batch
  int d = ((wid & 127) << 1) | (lane >> 4);            // 2 channels per wave
  float Aa = -__expf(A_log[d * DSZ + s]);
  float Dd = Dp[d];
  float h = 0.f;
  for (int l = 0; l < LSEQ; ++l) {
    size_t t = (size_t)b * LSEQ + l;
    float dlt = delta[t * DI + d];                     // broadcast in 16-group
    float u = xc[t * DI + d];
    float Bs = proj[t * NO + DRK + s];
    float Cs = proj[t * NO + DRK + DSZ + s];
    h = __expf(dlt * Aa) * h + (dlt * u) * Bs;
    float y = h * Cs;
    y += __shfl_xor(y, 1, 32);
    y += __shfl_xor(y, 2, 32);
    y += __shfl_xor(y, 4, 32);
    y += __shfl_xor(y, 8, 32);
    if (s == 0) {
      float yo = 0.7f * (y + u * Dd) + 0.3f * u;       // ALPHA = 0.3
      float g = xz[t * (2 * DI) + DI + d];             // gate half
      float sg = g / (1.0f + __expf(-g));
      y16[t * DI + d] = (_Float16)(yo * sg);
    }
  }
}

// ---------------------------------------------------------------------------
// Head: mean over L, layernorm(128), classifier (128 -> 10). Block per batch.
// ---------------------------------------------------------------------------
__global__ void head_kernel(const float* __restrict__ h, const float* __restrict__ nw,
                            const float* __restrict__ nb, const float* __restrict__ cw,
                            const float* __restrict__ cb, float* __restrict__ out) {
  __shared__ float pooled[DM];
  __shared__ float red[DM];
  int b = blockIdx.x;
  int m = threadIdx.x;
  const float* hb = h + (size_t)b * LSEQ * DM;
  float acc = 0.f;
  for (int l = 0; l < LSEQ; ++l) acc += hb[(size_t)l * DM + m];
  float p = acc * (1.0f / LSEQ);
  red[m] = p;
  __syncthreads();
  for (int off = 64; off; off >>= 1) { if (m < off) red[m] += red[m + off]; __syncthreads(); }
  float mean = red[0] * (1.0f / DM);
  __syncthreads();
  float dd = p - mean;
  red[m] = dd * dd;
  __syncthreads();
  for (int off = 64; off; off >>= 1) { if (m < off) red[m] += red[m + off]; __syncthreads(); }
  float var = red[0] * (1.0f / DM);
  pooled[m] = dd * rsqrtf(var + 1e-5f) * nw[m] + nb[m];
  __syncthreads();
  if (m < NCLS) {
    float o = cb[m];
    for (int k = 0; k < DM; ++k) o += pooled[k] * cw[k * NCLS + m];
    out[b * NCLS + m] = o;
  }
}

// ---------------------------------------------------------------------------
// Orchestration
// ---------------------------------------------------------------------------
extern "C" void kernel_launch(void* const* d_in, const int* in_sizes, int n_in,
                              void* d_out, int out_size, void* d_ws, size_t ws_size,
                              hipStream_t stream) {
  const float* x          = (const float*)d_in[0];
  const float* W_in       = (const float*)d_in[1];
  const float* b_in       = (const float*)d_in[2];
  const float* ln_w       = (const float*)d_in[3];
  const float* ln_b       = (const float*)d_in[4];
  const float* in_proj_w  = (const float*)d_in[5];
  const float* conv_w     = (const float*)d_in[6];
  const float* conv_b     = (const float*)d_in[7];
  const float* x_proj_w   = (const float*)d_in[8];
  const float* dt_proj_w  = (const float*)d_in[9];
  const float* dt_proj_b  = (const float*)d_in[10];
  const float* A_log      = (const float*)d_in[11];
  const float* Dp         = (const float*)d_in[12];
  const float* out_proj_w = (const float*)d_in[13];
  const float* out_proj_b = (const float*)d_in[14];
  const float* norm_w     = (const float*)d_in[15];
  const float* norm_b     = (const float*)d_in[16];
  const float* cls_w      = (const float*)d_in[17];
  const float* cls_b      = (const float*)d_in[18];
  float* out = (float*)d_out;

  // ---- workspace layout (bytes), ~190 MB total ----
  char* ws = (char*)d_ws;
  size_t off = 0;
  auto alloc = [&](size_t bytes) { char* p = ws + off; off += (bytes + 255) & ~(size_t)255; return p; };
  float*     hA    = (float*)alloc((size_t)BL * DM * 4);       // 16 MB
  float*     hB    = (float*)alloc((size_t)BL * DM * 4);       // 16 MB
  _Float16*  z16   = (_Float16*)alloc((size_t)BL * DM * 2);    //  8 MB
  float*     xz    = (float*)alloc((size_t)BL * 2 * DI * 4);   // 64 MB
  float*     xc    = (float*)alloc((size_t)BL * DI * 4);       // 32 MB
  float*     proj  = (float*)alloc((size_t)BL * 40 * 4);       //  5 MB
  float*     delta = (float*)alloc((size_t)BL * DI * 4);       // 32 MB
  _Float16*  y16   = (_Float16*)alloc((size_t)BL * DI * 2);    // 16 MB
  _Float16*  w1p   = (_Float16*)alloc((size_t)NLAY * DM * 2 * DI * 2);  // in_proj packed
  _Float16*  w2p   = (_Float16*)alloc((size_t)NLAY * DI * DM * 2);      // out_proj packed
  (void)ws_size; (void)in_sizes; (void)n_in; (void)out_size;

  // ---- pack weights into WMMA B-matrix layout (all layers at once) ----
  {
    int tot1 = NLAY * DM * 2 * DI;       // 262144
    pack_w_kernel<<<(tot1 + 255) / 256, 256, 0, stream>>>(in_proj_w, w1p, DM, 2 * DI, tot1);
    int tot2 = NLAY * DI * DM;           // 131072
    pack_w_kernel<<<(tot2 + 255) / 256, 256, 0, stream>>>(out_proj_w, w2p, DI, DM, tot2);
  }

  // ---- embed ----
  embed_kernel<<<(BL * DM) / 256, 256, 0, stream>>>(x, W_in, b_in, hA);

  // ---- layers ----
  for (int i = 0; i < NLAY; ++i) {
    float* hin  = (i & 1) ? hB : hA;
    float* hout = (i & 1) ? hA : hB;

    // layernorm -> f16
    ln_to_f16_kernel<<<(BL * 32) / 256, 256, 0, stream>>>(hin, ln_w + i * DM, ln_b + i * DM, z16);

    // xz = z @ in_proj_w      (M=32768, N=512, K=128) — WMMA, 16x64 per wave
    {
      int tiles = (BL / 16) * ((2 * DI) / 64);    // 16384 waves
      wmma_gemm_kernel<<<tiles / 8, 256, 0, stream>>>(
          z16, w1p + (size_t)i * DM * 2 * DI, xz, nullptr, nullptr,
          BL, 2 * DI, DM, 0);
    }

    // depthwise causal conv + SiLU
    conv_silu_kernel<<<(BL * DI) / 256, 256, 0, stream>>>(
        xz, conv_w + i * DI * DCV, conv_b + i * DI, xc);

    // x_proj (K=256 -> 40)
    {
      int tot = BL * (DRK + 2 * DSZ);
      xproj_kernel<<<(tot + 255) / 256, 256, 0, stream>>>(xc, x_proj_w + (size_t)i * DI * 40, proj);
    }

    // delta = softplus(dt @ dt_proj_w + b)
    delta_kernel<<<(BL * DI) / 256, 256, 0, stream>>>(
        proj, dt_proj_w + (size_t)i * DRK * DI, dt_proj_b + i * DI, delta);

    // selective scan + skip + observer blend + gate -> y16
    // 4096 waves: 512 blocks x 256 threads (8 waves each)
    ssm_kernel<<<512, 256, 0, stream>>>(
        delta, proj, xc, xz, A_log + (size_t)i * DI * DSZ, Dp + i * DI, y16);

    // h = res + y @ out_proj_w + out_proj_b   (M=32768, N=128, K=256) — WMMA fused
    {
      int tiles = (BL / 16) * (DM / 64);          // 4096 waves
      wmma_gemm_kernel<<<tiles / 8, 256, 0, stream>>>(
          y16, w2p + (size_t)i * DI * DM, hout, hin, out_proj_b + i * DM,
          BL, DM, DI, 1);
    }
  }

  // after 4 layers final h is in hA (A->B->A->B->A)
  head_kernel<<<BATCH, DM, 0, stream>>>(hA, norm_w, norm_b, cls_w, cls_b, out);
}
